// Model2_3925600109170
// MI455X (gfx1250) — matrix operand order repaired and checked by reading, not compiled, not run on hardware
//
#include <hip/hip_runtime.h>
#include <hip/hip_bf16.h>

#define N_NEWS 100000
#define D_NEWS 128
#define HDIM   64
#define E_LINKS 1600000
#define QN     1024

typedef __attribute__((ext_vector_type(16))) _Float16 v16h;
typedef __attribute__((ext_vector_type(8)))  float    v8f;

// ---------------- float atomic max via integer ordering trick ----------------
__device__ __forceinline__ void atomicMaxFloat(float* addr, float val) {
    if (val >= 0.0f) {
        atomicMax((int*)addr, __float_as_int(val));
    } else {
        atomicMin((unsigned int*)addr, __float_as_uint(val));
    }
}

// ---------------- init workspace: qof=-1, m=-inf, den=0, num=0 ----------------
__global__ void k_init(int* __restrict__ qof, float* __restrict__ m,
                       float* __restrict__ den, float* __restrict__ num) {
    int t = blockIdx.x * blockDim.x + threadIdx.x;
    if (t < N_NEWS) qof[t] = -1;
    if (t < QN) { m[t] = __int_as_float(0xFF800000); den[t] = 0.0f; }
    if (t < QN * HDIM) num[t] = 0.0f;
}

// ---------------- scatter query slots ----------------
__global__ void k_scatter(const int* __restrict__ nidx, int* __restrict__ qof) {
    int q = blockIdx.x * blockDim.x + threadIdx.x;
    if (q < QN) qof[nidx[q]] = q;   // duplicate dsts share a slot: result identical
}

// ---------------- one-shot weight convert: wh = f16(gat_n_ws) [64x128] ----------
__global__ void k_cvt_w(const float* __restrict__ w, _Float16* __restrict__ wh) {
    int t = blockIdx.x * blockDim.x + threadIdx.x;   // 8192 elements
    if (t < HDIM * D_NEWS) wh[t] = (_Float16)w[t];
}

// ---------------- vs = a_s @ ws, vd = a_d @ wd (fold attention vectors) -------
__global__ void k_vsvd(const float* __restrict__ ws, const float* __restrict__ wd,
                       const float* __restrict__ as_, const float* __restrict__ ad_,
                       float* __restrict__ vs, float* __restrict__ vd) {
    int k = threadIdx.x;                  // 128 threads
    float s = 0.0f, d = 0.0f;
    for (int h = 0; h < HDIM; ++h) {
        s += as_[h] * ws[h * D_NEWS + k];
        d += ad_[h] * wd[h * D_NEWS + k];
    }
    vs[k] = s; vd[k] = d;
}

// ---------------- es/ed: per-node 128-dim dot products ----------------
__global__ void k_esed(const float* __restrict__ x, const float* __restrict__ vs,
                       const float* __restrict__ vd, float* __restrict__ es,
                       float* __restrict__ ed) {
    int i = blockIdx.x * blockDim.x + threadIdx.x;
    if (i >= N_NEWS) return;
    const float4* xr  = (const float4*)(x + (size_t)i * D_NEWS);
    const float4* vs4 = (const float4*)vs;
    const float4* vd4 = (const float4*)vd;
    float s = 0.0f, d = 0.0f;
    #pragma unroll
    for (int g = 0; g < D_NEWS / 4; ++g) {
        float4 p = xr[g], a = vs4[g], b = vd4[g];
        s += p.x * a.x + p.y * a.y + p.z * a.z + p.w * a.w;
        d += p.x * b.x + p.y * b.y + p.z * b.z + p.w * b.w;
    }
    es[i] = s; ed[i] = d;
}

// ---------------- WMMA GEMM: hs[100000,64] = x[100000,128] @ ws^T ----------------
// One wave32 computes a 16x64 tile: 4 f32 accumulators, 16 v_wmma_f32_16x16x32_f16.
// A: f32 loads + in-register cvt (32 cvt_pk/lane/tile). B: pre-converted f16
// weights, each fragment = 16 contiguous halves = two global_load_b128, no cvt.
__global__ void k_gemm_hs(const float* __restrict__ x, const _Float16* __restrict__ wh,
                          float* __restrict__ hs) {
    const int lane = threadIdx.x & 31;
    const int wave = threadIdx.x >> 5;
    const int tile = blockIdx.x * 8 + wave;        // wave-uniform
    if (tile * 16 >= N_NEWS) return;               // whole wave exits -> EXEC all-1s
    const int m0 = tile * 16;
    const int hi = lane >> 4;                      // 0 or 1
    const int lo = lane & 15;
    const float4* xr = (const float4*)(x + (size_t)(m0 + lo) * D_NEWS);

    v8f c[4] = {v8f{}, v8f{}, v8f{}, v8f{}};

    #pragma unroll
    for (int kt = 0; kt < 4; ++kt) {
        const int k0 = kt * 32;
        // A fragment (16x32 f16): lane(hi,lo) -> M=lo, K = k0+16hi+j
        v16h a;
        {
            int g = (k0 + hi * 16) >> 2;
            float4 p0 = xr[g], p1 = xr[g + 1];
            float4 p2 = xr[g + 2], p3 = xr[g + 3];
            a[0]=(_Float16)p0.x; a[1]=(_Float16)p0.y; a[2]=(_Float16)p0.z; a[3]=(_Float16)p0.w;
            a[4]=(_Float16)p1.x; a[5]=(_Float16)p1.y; a[6]=(_Float16)p1.z; a[7]=(_Float16)p1.w;
            a[8]=(_Float16)p2.x; a[9]=(_Float16)p2.y; a[10]=(_Float16)p2.z; a[11]=(_Float16)p2.w;
            a[12]=(_Float16)p3.x; a[13]=(_Float16)p3.y; a[14]=(_Float16)p3.z; a[15]=(_Float16)p3.w;
        }
        #pragma unroll
        for (int nt = 0; nt < 4; ++nt) {
            // B fragment (32x16 f16): lane(hi,lo) -> N=nt*16+lo, K = k0+16hi+j
            // 16 contiguous halves, 32B aligned -> single b128 load pair, no cvt.
            const v16h* wp = (const v16h*)(wh + (size_t)(nt * 16 + lo) * D_NEWS + k0 + hi * 16);
            v16h b = *wp;
            c[nt] = __builtin_amdgcn_wmma_f32_16x16x32_f16(false, a, false, b,
                                                           (short)0, c[nt], false, false);
        }
    }
    // D layout: VGPR r, lanes<16 -> M=r, lanes>=16 -> M=r+8; N = lane&15
    #pragma unroll
    for (int nt = 0; nt < 4; ++nt)
        #pragma unroll
        for (int r = 0; r < 8; ++r)
            hs[(size_t)(m0 + hi * 8 + r) * HDIM + nt * 16 + lo] = c[nt][r];
}

// ---------------- edge pass 1: segment max over filtered edges ----------------
__global__ void k_edge_max(const int* __restrict__ src, const int* __restrict__ dst,
                           const int* __restrict__ qof, const float* __restrict__ es,
                           const float* __restrict__ ed, float* __restrict__ m) {
    int t = blockIdx.x * blockDim.x + threadIdx.x;
    if (t >= E_LINKS) return;
    int d = dst[t];
    int slot = qof[d];
    if (slot < 0) return;                 // ~99% of edges filtered here
    float e = es[src[t]] + ed[d];
    e = e > 0.0f ? e : 0.2f * e;          // leaky_relu(0.2)
    atomicMaxFloat(m + slot, e);
}

// ---------------- edge pass 2: softmax-weighted accumulation ----------------
__global__ void k_edge_acc(const int* __restrict__ src, const int* __restrict__ dst,
                           const int* __restrict__ qof, const float* __restrict__ es,
                           const float* __restrict__ ed, const float* __restrict__ m,
                           const float* __restrict__ hs, float* __restrict__ den,
                           float* __restrict__ num) {
    int t = blockIdx.x * blockDim.x + threadIdx.x;
    if (t >= E_LINKS) return;
    int d = dst[t];
    int slot = qof[d];
    if (slot < 0) return;
    int s = src[t];
    float e = es[s] + ed[d];
    e = e > 0.0f ? e : 0.2f * e;
    float wgt = expf(e - m[slot]);
    atomicAdd(den + slot, wgt);
    const float4* hr = (const float4*)(hs + (size_t)s * HDIM);
    float* np = num + (size_t)slot * HDIM;
    #pragma unroll
    for (int g = 0; g < HDIM / 4; ++g) {
        float4 p = hr[g];
        atomicAdd(np + g * 4 + 0, wgt * p.x);
        atomicAdd(np + g * 4 + 1, wgt * p.y);
        atomicAdd(np + g * 4 + 2, wgt * p.z);
        atomicAdd(np + g * 4 + 3, wgt * p.w);
    }
}

// ---------------- MLP head on the 1024 queried rows ----------------
__global__ void k_head(const int* __restrict__ nidx, const int* __restrict__ qof,
                       const float* __restrict__ num, const float* __restrict__ den,
                       const float* __restrict__ gb, const float* __restrict__ l1w,
                       const float* __restrict__ l1b, const float* __restrict__ l2w,
                       const float* __restrict__ l2b, float* __restrict__ out) {
    __shared__ float sh[HDIM];
    __shared__ float sx[HDIM];
    int q = blockIdx.x;
    int j = threadIdx.x;                  // 64 threads
    int slot = qof[nidx[q]];              // always >= 0 for queried nodes
    float dn = den[slot];
    dn = dn > 1e-16f ? dn : 1e-16f;
    sh[j] = num[slot * HDIM + j] / dn + gb[j];
    __syncthreads();
    float acc = l1b[j];
    #pragma unroll
    for (int k = 0; k < HDIM; ++k) acc += l1w[j * HDIM + k] * sh[k];
    sx[j] = acc > 0.0f ? acc : 0.0f;
    __syncthreads();
    if (j < 32) {
        float o = l2b[j];
        #pragma unroll
        for (int k = 0; k < HDIM; ++k) o += l2w[j * HDIM + k] * sx[k];
        out[q * 32 + j] = o;
    }
}

extern "C" void kernel_launch(void* const* d_in, const int* in_sizes, int n_in,
                              void* d_out, int out_size, void* d_ws, size_t ws_size,
                              hipStream_t stream) {
    (void)in_sizes; (void)n_in; (void)out_size; (void)ws_size;
    const float* x_news       = (const float*)d_in[0];
    const float* gat_n_ws     = (const float*)d_in[11];
    const float* gat_n_wd     = (const float*)d_in[12];
    const float* gat_n_as     = (const float*)d_in[13];
    const float* gat_n_ad     = (const float*)d_in[14];
    const float* gat_n_b      = (const float*)d_in[15];
    const float* lin1_w       = (const float*)d_in[16];
    const float* lin1_b       = (const float*)d_in[17];
    const float* lin2_w       = (const float*)d_in[18];
    const float* lin2_b       = (const float*)d_in[19];
    const int*   links_src    = (const int*)d_in[24];
    const int*   links_dst    = (const int*)d_in[25];
    const int*   news_indices = (const int*)d_in[27];
    float* out = (float*)d_out;

    // carve workspace (~27 MB)
    char* p = (char*)d_ws;
    auto carve = [&](size_t bytes) { void* r = (void*)p; p += (bytes + 255) & ~(size_t)255; return r; };
    float*     hs  = (float*)carve((size_t)N_NEWS * HDIM * 4);
    float*     es  = (float*)carve((size_t)N_NEWS * 4);
    float*     ed  = (float*)carve((size_t)N_NEWS * 4);
    float*     vs  = (float*)carve(D_NEWS * 4);
    float*     vd  = (float*)carve(D_NEWS * 4);
    int*       qof = (int*)  carve((size_t)N_NEWS * 4);
    float*     m   = (float*)carve(QN * 4);
    float*     den = (float*)carve(QN * 4);
    float*     num = (float*)carve((size_t)QN * HDIM * 4);
    _Float16*  wh  = (_Float16*)carve((size_t)HDIM * D_NEWS * 2);

    k_init<<<(N_NEWS + 255) / 256, 256, 0, stream>>>(qof, m, den, num);
    k_scatter<<<(QN + 255) / 256, 256, 0, stream>>>(news_indices, qof);
    k_cvt_w<<<(HDIM * D_NEWS + 255) / 256, 256, 0, stream>>>(gat_n_ws, wh);
    k_vsvd<<<1, D_NEWS, 0, stream>>>(gat_n_ws, gat_n_wd, gat_n_as, gat_n_ad, vs, vd);
    k_esed<<<(N_NEWS + 255) / 256, 256, 0, stream>>>(x_news, vs, vd, es, ed);
    // 6250 row tiles, 8 waves/block
    k_gemm_hs<<<(N_NEWS / 16 + 7) / 8, 256, 0, stream>>>(x_news, wh, hs);
    k_edge_max<<<(E_LINKS + 255) / 256, 256, 0, stream>>>(links_src, links_dst, qof, es, ed, m);
    k_edge_acc<<<(E_LINKS + 255) / 256, 256, 0, stream>>>(links_src, links_dst, qof, es, ed, m, hs, den, num);
    k_head<<<QN, HDIM, 0, stream>>>(news_indices, qof, num, den, gat_n_b, lin1_w, lin1_b, lin2_w, lin2_b, out);
}
